// CTLSTM_60464549593103
// MI455X (gfx1250) — matrix operand order repaired and checked by reading, not compiled, not run on hardware
//
#include <hip/hip_runtime.h>

typedef unsigned short u16;
typedef unsigned int   u32;
typedef __attribute__((ext_vector_type(16))) __bf16 v16bf;
typedef __attribute__((ext_vector_type(8)))  float  v8f;

#define KDIM  2048
#define NEV   4096
#define NSTEP 4095

// ---------- bf16 helpers (RNE) ----------
__device__ __forceinline__ u16 f2bf(float f) {
    u32 u = __float_as_uint(f);
    u32 r = u + 0x7FFFu + ((u >> 16) & 1u);
    return (u16)(r >> 16);
}
__device__ __forceinline__ float bf2f(u32 lo16) {
    return __uint_as_float(lo16 << 16);
}

// =====================================================================
// Phase A0: convert weight sub-blocks to bf16 (once per launch, amortized
// over 4095 reuses; halves scan bandwidth and makes the whole recurrent
// working set (40MB) L2-resident on MI455X's 192MB L2).
// =====================================================================
__global__ __launch_bounds__(256) void conv_bf16(const float* __restrict__ src,
                                                 u16* __restrict__ dst,
                                                 int total, int cols, int stride, int coff) {
    int i = blockIdx.x * 256 + threadIdx.x;
    if (i < total) {
        int r = i / cols, c = i - r * cols;
        dst[i] = f2bf(src[(size_t)r * stride + coff + c]);
    }
}

// =====================================================================
// Phase A1: input-part gate GEMM via WMMA bf16.
//   out[m, n] = bias[n] + sum_k seq[m, k] * W[n, k]   (k in [0, K))
// Block tile 128x128, K-tile 64 (two 32-deep WMMA sub-steps per stage:
// 16 v_wmma per barrier pair), 8 waves (wave32), each wave 2x4 16x16 accs.
// =====================================================================
__global__ __launch_bounds__(256) void gemm_in_wmma(const float* __restrict__ A,   // [4096 x 2048] seq
                                                    const float* __restrict__ W,   // [2048 x 4096] gate weight
                                                    const float* __restrict__ bias,// [2048]
                                                    float* __restrict__ out)       // [4096 x 2048]
{
    // LDS tiles in bf16: 128 rows x 64 K, row stride 72 ushorts
    // (144B = 9*16B: every fragment offset stays 16B-aligned, banks spread)
    __shared__ u16 As[128 * 72];
    __shared__ u16 Bs[128 * 72];

    const int tid  = threadIdx.x;
    const int lane = tid & 31;
    const int wv   = tid >> 5;      // 0..7
    const int wm   = wv & 3;        // 32-row M quadrant
    const int wn   = wv >> 2;       // 64-col N half
    const int m0   = blockIdx.y * 128;
    const int n0   = blockIdx.x * 128;

    v8f acc[2][4];
    for (int mi = 0; mi < 2; ++mi)
        for (int ni = 0; ni < 4; ++ni)
            for (int e = 0; e < 8; ++e) acc[mi][ni][e] = 0.f;

    union Frag { uint4 q[2]; v16bf v; };

    const int l15 = lane & 15;
    const int kbA = (lane >> 4) * 8;    // ushort offset in row (A frag)
    const int kbB = (lane >> 4) * 8;    // ushort offset in row (B frag)

    for (int kk = 0; kk < KDIM / 64; ++kk) {
        const int k0 = kk * 64;
        // ---- stage f32 -> bf16 128x64 tiles into LDS (8 float4/thread/tile)
        #pragma unroll
        for (int j = 0; j < 8; ++j) {
            int fidx = tid * 8 + j;          // 0..2047
            int row = fidx >> 4, c4 = fidx & 15;
            float4 va = *(const float4*)&A[(size_t)(m0 + row) * KDIM + k0 + c4 * 4];
            uint2 pa;
            pa.x = (u32)f2bf(va.x) | ((u32)f2bf(va.y) << 16);
            pa.y = (u32)f2bf(va.z) | ((u32)f2bf(va.w) << 16);
            *(uint2*)&As[row * 72 + c4 * 4] = pa;
            float4 vb = *(const float4*)&W[(size_t)(n0 + row) * (2 * KDIM) + k0 + c4 * 4];
            uint2 pb;
            pb.x = (u32)f2bf(vb.x) | ((u32)f2bf(vb.y) << 16);
            pb.y = (u32)f2bf(vb.z) | ((u32)f2bf(vb.w) << 16);
            *(uint2*)&Bs[row * 72 + c4 * 4] = pb;
        }
        __syncthreads();

        // ---- two K=32 sub-steps from the staged tile
        #pragma unroll
        for (int ks = 0; ks < 2; ++ks) {
            const int ko = ks * 32;          // ushort offset of sub-tile
            Frag aF[2], bF[4];
            #pragma unroll
            for (int mi = 0; mi < 2; ++mi) {
                int row = wm * 32 + mi * 16 + l15;
                aF[mi].q[0] = *(const uint4*)&As[row * 72 + ko + kbA];       // e0..7 : K=kb..
                aF[mi].q[1] = *(const uint4*)&As[row * 72 + ko + 16 + kbA];  // e8..15: K=16+kb..
            }
            #pragma unroll
            for (int ni = 0; ni < 4; ++ni) {
                int row = wn * 64 + ni * 16 + l15;                           // N column
                bF[ni].q[0] = *(const uint4*)&Bs[row * 72 + ko + kbB];       // K=kb..kb+7
                bF[ni].q[1] = *(const uint4*)&Bs[row * 72 + ko + 16 + kbB];  // K=16+kb..
            }
            #pragma unroll
            for (int mi = 0; mi < 2; ++mi)
                #pragma unroll
                for (int ni = 0; ni < 4; ++ni)
                    acc[mi][ni] = __builtin_amdgcn_wmma_f32_16x16x32_bf16(
                        false, aF[mi].v, false, bF[ni].v, (short)0, acc[mi][ni], false, false);
        }
        __syncthreads();
    }

    // ---- epilogue: D VGPR v -> M = (lane>>4)*8 + v, N = lane&15
    const int mAdd = (lane >> 4) * 8;
    #pragma unroll
    for (int mi = 0; mi < 2; ++mi)
        #pragma unroll
        for (int ni = 0; ni < 4; ++ni)
            #pragma unroll
            for (int e = 0; e < 8; ++e) {
                int m = m0 + wm * 32 + mi * 16 + mAdd + e;
                int n = n0 + wn * 64 + ni * 16 + l15;
                out[(size_t)m * KDIM + n] = acc[mi][ni][e] + bias[n];
            }
}

// =====================================================================
// Phase A2: input part of scalar decay: GD[s] = bd + seq[s] . Wd[0:K]
// =====================================================================
__global__ __launch_bounds__(256) void gd_kernel(const float* __restrict__ seq,
                                                 const float* __restrict__ Wd,
                                                 const float* __restrict__ bd,
                                                 float* __restrict__ GD) {
    __shared__ float red[256];
    int s = blockIdx.x, t = threadIdx.x;
    float p = 0.f;
    for (int c = t; c < KDIM; c += 256) p += seq[(size_t)s * KDIM + c] * Wd[c];
    red[t] = p; __syncthreads();
    for (int o = 128; o; o >>= 1) { if (t < o) red[t] += red[t + o]; __syncthreads(); }
    if (t == 0) GD[s] = red[0] + bd[0];
}

// =====================================================================
// Phase A3: zero state + barrier counters (per-launch determinism)
// =====================================================================
__global__ __launch_bounds__(256) void init_state(float* ht0, float* ht1, float* ct, float* cb, u32* bar) {
    int t = blockIdx.x * 256 + threadIdx.x;
    if (t < KDIM) { ht0[t] = 0.f; ht1[t] = 0.f; ct[t] = 0.f; cb[t] = 0.f; }
    if (t < 2) bar[t] = 0u;
}

// =====================================================================
// Phase B: persistent sequential scan.
// 64 blocks x 512 threads; block b owns state rows [b*32, b*32+32).
// Per step: (1) 4 gate hidden matvecs (bf16 weights, L2-resident) +
// redundant scalar-decay dot, elementwise update, write ht_new;
// (2) one device-wide barrier; (3) Wl matvec on ht_new -> softplus out.
// ht double-buffered so one barrier/step suffices.
// =====================================================================
__device__ __forceinline__ float dot_row_bf16(const uint4* __restrict__ w4,
                                              const float* shHt, int lane) {
    float acc = 0.f;
    #pragma unroll
    for (int jj = 0; jj < 8; ++jj) {
        uint4 wvv = w4[jj * 32 + lane];
        const float* hp = &shHt[(jj * 32 + lane) * 8];
        float4 h0 = *(const float4*)hp;
        float4 h1 = *(const float4*)(hp + 4);
        acc += bf2f(wvv.x & 0xFFFFu) * h0.x + bf2f(wvv.x >> 16) * h0.y;
        acc += bf2f(wvv.y & 0xFFFFu) * h0.z + bf2f(wvv.y >> 16) * h0.w;
        acc += bf2f(wvv.z & 0xFFFFu) * h1.x + bf2f(wvv.z >> 16) * h1.y;
        acc += bf2f(wvv.w & 0xFFFFu) * h1.z + bf2f(wvv.w >> 16) * h1.w;
    }
    return acc;
}

__global__ __launch_bounds__(512)
void ctlstm_scan(const float* __restrict__ G,     // [4][4096][2048] gate input parts (+bias)
                 const float* __restrict__ GD,    // [4096]
                 const u16*   __restrict__ Wh,    // [4][2048][2048] bf16 hidden weights
                 const u16*   __restrict__ Wdh,   // [2048] bf16 hidden decay row
                 const u16*   __restrict__ Wlb,   // [2048][2048] bf16 Wl
                 const float* __restrict__ times, // [4096]
                 const float* __restrict__ scale, // [2048]
                 float* ct, float* cbar, float* ht0, float* ht1,
                 u32* bar, float* __restrict__ out) {
    __shared__ float shHt[KDIM];
    __shared__ float shGate[4][32];
    __shared__ float shRed[512];
    __shared__ float shDelta, shDt;

    const int tid  = threadIdx.x;
    const int lane = tid & 31;
    const int wv   = tid >> 5;            // 0..15
    const int rb   = blockIdx.x * 32;
    u32* cnt = bar; u32* gen = bar + 1;
    const u32 NBLK = gridDim.x;

    for (int s = 0; s < NSTEP; ++s) {
        const float* htOld = (s & 1) ? ht1 : ht0;
        float*       htNew = (s & 1) ? ht0 : ht1;

        // ---- stage ht into LDS ----
        for (int c = tid; c < KDIM; c += 512) shHt[c] = htOld[c];
        __syncthreads();

        // ---- scalar decay (computed redundantly per block; 2K MACs) ----
        float p = 0.f;
        for (int c = tid; c < KDIM; c += 512) p += shHt[c] * bf2f(Wdh[c]);
        shRed[tid] = p; __syncthreads();
        for (int o = 256; o; o >>= 1) { if (tid < o) shRed[tid] += shRed[tid + o]; __syncthreads(); }
        if (tid == 0) {
            float d = GD[s] + shRed[0];
            shDelta = d > 0.f ? d : 0.f;
            shDt = times[s + 1] - (s ? times[s] : 0.f);
        }
        __syncthreads();

        // ---- 4 gate hidden matvecs: 128 row-dots / block, 8 per wave ----
        #pragma unroll 1
        for (int k = 0; k < 8; ++k) {
            int d  = wv * 8 + k;          // 0..127
            int g  = d >> 5;              // gate index
            int lr = d & 31;              // local row
            int r  = rb + lr;
            const uint4* w4 = (const uint4*)(Wh + ((size_t)g * KDIM + r) * KDIM);
            float acc = dot_row_bf16(w4, shHt, lane);
            for (int o = 16; o; o >>= 1) acc += __shfl_xor(acc, o);
            if (lane == 0) shGate[g][lr] = acc;
        }
        __syncthreads();

        // ---- elementwise state update (32 owned rows) ----
        if (tid < 32) {
            int r = rb + tid;
            const size_t MN = (size_t)NEV * KDIM;
            size_t base = (size_t)s * KDIM + r;
            float iv = G[0 * MN + base] + shGate[0][tid]; iv = iv > 0.f ? iv : 0.f;
            float fv = G[1 * MN + base] + shGate[1][tid]; fv = fv > 0.f ? fv : 0.f;
            float zv = G[2 * MN + base] + shGate[2][tid]; zv = zv > 0.f ? zv : 0.f;
            float ov = G[3 * MN + base] + shGate[3][tid]; ov = ov > 0.f ? ov : 0.f;
            float c0 = ct[r], cb = cbar[r];
            float iz   = iv * zv;
            float clow = fv * c0 + iz;
            float cbn  = fv * cb + iz;
            float e    = __expf(shDt * shDelta);
            float ctn  = cbn + (clow - cbn) * e;
            float t2   = 2.f * ctn; t2 = t2 > 0.f ? t2 : 0.f;
            float htn  = ov * (2.f * t2 - 1.f);
            ct[r] = ctn; cbar[r] = cbn; htNew[r] = htn;
        }

        // ---- device-wide split barrier (ht_new complete) ----
        __threadfence();
        __syncthreads();
        if (tid == 0) {
            u32 g0 = __hip_atomic_load(gen, __ATOMIC_ACQUIRE, __HIP_MEMORY_SCOPE_AGENT);
            u32 a  = __hip_atomic_fetch_add(cnt, 1u, __ATOMIC_ACQ_REL, __HIP_MEMORY_SCOPE_AGENT);
            if (a == NBLK - 1u) {
                __hip_atomic_store(cnt, 0u, __ATOMIC_RELAXED, __HIP_MEMORY_SCOPE_AGENT);
                __hip_atomic_fetch_add(gen, 1u, __ATOMIC_RELEASE, __HIP_MEMORY_SCOPE_AGENT);
            } else {
                while (__hip_atomic_load(gen, __ATOMIC_ACQUIRE, __HIP_MEMORY_SCOPE_AGENT) == g0)
                    __builtin_amdgcn_s_sleep(1);
            }
        }
        __syncthreads();
        __threadfence();

        // ---- intensities: lamb = scale * softplus((ht_new . Wl[r]) / scale) ----
        for (int c = tid; c < KDIM; c += 512) shHt[c] = htNew[c];
        __syncthreads();
        #pragma unroll
        for (int q = 0; q < 2; ++q) {
            int r = rb + wv * 2 + q;
            const uint4* w4 = (const uint4*)(Wlb + (size_t)r * KDIM);
            float acc = dot_row_bf16(w4, shHt, lane);
            for (int o = 16; o; o >>= 1) acc += __shfl_xor(acc, o);
            if (lane == 0) {
                float sc = scale[r];
                float x  = acc / sc;
                float sp = (x > 20.f) ? x : log1pf(__expf(x));
                out[(size_t)s * KDIM + r] = sc * sp;
            }
        }
        __syncthreads();   // protect shHt before next step's reload
    }
}

// =====================================================================
// Host-side launcher
// =====================================================================
extern "C" void kernel_launch(void* const* d_in, const int* in_sizes, int n_in,
                              void* d_out, int out_size, void* d_ws, size_t ws_size,
                              hipStream_t stream) {
    (void)in_sizes; (void)n_in; (void)out_size; (void)ws_size;
    const float* seq   = (const float*)d_in[0];
    const float* times = (const float*)d_in[1];
    const float* Wi    = (const float*)d_in[2];
    const float* bi    = (const float*)d_in[3];
    const float* Wf    = (const float*)d_in[4];
    const float* bf_   = (const float*)d_in[5];
    const float* Wz    = (const float*)d_in[6];
    const float* bz    = (const float*)d_in[7];
    const float* Wo    = (const float*)d_in[8];
    const float* bo    = (const float*)d_in[9];
    const float* Wd    = (const float*)d_in[10];
    const float* bd    = (const float*)d_in[11];
    const float* Wl    = (const float*)d_in[12];
    const float* scale = (const float*)d_in[13];
    float* out = (float*)d_out;

    // ---- workspace carve-out (256B aligned) ----
    char* wsb = (char*)d_ws;
    size_t off = 0;
    auto alloc = [&](size_t bytes) -> void* {
        void* pp = wsb + off;
        off = (off + bytes + 255) & ~(size_t)255;
        return pp;
    };
    const size_t MN = (size_t)NEV * KDIM;
    float* G    = (float*)alloc(4 * MN * sizeof(float));              // 128 MB
    float* GD   = (float*)alloc(NEV * sizeof(float));
    u16*   WhA  = (u16*)alloc(4 * (size_t)KDIM * KDIM * sizeof(u16)); // 32 MB
    u16*   Wlb  = (u16*)alloc((size_t)KDIM * KDIM * sizeof(u16));     // 8 MB
    u16*   Wdh  = (u16*)alloc(KDIM * sizeof(u16));
    float* ht0  = (float*)alloc(KDIM * sizeof(float));
    float* ht1  = (float*)alloc(KDIM * sizeof(float));
    float* ctv  = (float*)alloc(KDIM * sizeof(float));
    float* cbv  = (float*)alloc(KDIM * sizeof(float));
    u32*   bar  = (u32*)alloc(256);

    // ---- A0: bf16 weight conversion (hidden halves + Wl + Wd hidden) ----
    const int WT = KDIM * KDIM;                 // 4,194,304
    const int CB = (WT + 255) / 256;
    conv_bf16<<<CB, 256, 0, stream>>>(Wi, WhA + 0 * (size_t)WT, WT, KDIM, 2 * KDIM, KDIM);
    conv_bf16<<<CB, 256, 0, stream>>>(Wf, WhA + 1 * (size_t)WT, WT, KDIM, 2 * KDIM, KDIM);
    conv_bf16<<<CB, 256, 0, stream>>>(Wz, WhA + 2 * (size_t)WT, WT, KDIM, 2 * KDIM, KDIM);
    conv_bf16<<<CB, 256, 0, stream>>>(Wo, WhA + 3 * (size_t)WT, WT, KDIM, 2 * KDIM, KDIM);
    conv_bf16<<<CB, 256, 0, stream>>>(Wl, Wlb, WT, KDIM, KDIM, 0);
    conv_bf16<<<(KDIM + 255) / 256, 256, 0, stream>>>(Wd, Wdh, KDIM, KDIM, 2 * KDIM, KDIM);

    // ---- A1: input-part gate GEMMs (WMMA bf16) ----
    dim3 gg(KDIM / 128, NEV / 128);   // (16, 32)
    gemm_in_wmma<<<gg, 256, 0, stream>>>(seq, Wi, bi,  G + 0 * MN);
    gemm_in_wmma<<<gg, 256, 0, stream>>>(seq, Wf, bf_, G + 1 * MN);
    gemm_in_wmma<<<gg, 256, 0, stream>>>(seq, Wz, bz,  G + 2 * MN);
    gemm_in_wmma<<<gg, 256, 0, stream>>>(seq, Wo, bo,  G + 3 * MN);

    // ---- A2/A3 ----
    gd_kernel<<<NEV, 256, 0, stream>>>(seq, Wd, bd, GD);
    init_state<<<(KDIM + 255) / 256, 256, 0, stream>>>(ht0, ht1, ctv, cbv, bar);

    // ---- B: persistent sequential scan ----
    ctlstm_scan<<<64, 512, 0, stream>>>(G, GD, WhA, Wdh, Wlb, times, scale,
                                        ctv, cbv, ht0, ht1, bar, out);
}
